// Qwen3NextSparseMoeBlock_89678917140578
// MI455X (gfx1250) — matrix operand order repaired and checked by reading, not compiled, not run on hardware
//
#include <hip/hip_runtime.h>
#include <hip/hip_bf16.h>
#include <math.h>

// Problem dims (fixed by the reference)
#define T_TOK 2048
#define H_DIM 2048
#define E_EXP 64
#define F_DIM 512
#define K_TOP 8

typedef __bf16 bf16;
typedef __attribute__((ext_vector_type(16))) __bf16 v16bf;
typedef __attribute__((ext_vector_type(4)))  __bf16 v4bf;
typedef __attribute__((ext_vector_type(8)))  float  v8f;

// ---------------------------------------------------------------------------
// WMMA fragment helpers (contiguous k order, same in A and B, wave32)
// ---------------------------------------------------------------------------
// A-matrix 16x32 bf16: lane L(0-15) -> row M=L, elems 0..15 = K 0..15
//                      lane L(16-31)-> row M=L-16, elems 0..15 = K 16..31
__device__ __forceinline__ v16bf load_a_frag(const bf16* __restrict__ base, int ld, int lane) {
  const int row = lane & 15;
  const int kb  = (lane >> 4) << 4;           // 0 or 16
  const bf16* p = base + row * ld + kb;
  v16bf a;
#pragma unroll
  for (int i = 0; i < 8; ++i) a[i] = p[i];
#pragma unroll
  for (int i = 0; i < 8; ++i) a[i + 8] = p[8 + i];
  return a;
}

// B-matrix 32x16 bf16 from row-major fp32 W[k][n] (convert in-register):
// lane L -> col N = L&15 ; elems 0..15 = K = (L>>4)*16 + i
__device__ __forceinline__ v16bf load_b_frag(const float* __restrict__ W, int ldw,
                                             int kbase, int nbase, int lane) {
  const float* p = W + (size_t)(kbase + ((lane >> 4) << 4)) * ldw + (nbase + (lane & 15));
  v16bf b;
#pragma unroll
  for (int i = 0; i < 16; ++i) b[i] = (bf16)p[(size_t)i * ldw];
  return b;
}

__device__ __forceinline__ v8f wmma_bf16(v16bf a, v16bf b, v8f c) {
  return __builtin_amdgcn_wmma_f32_16x16x32_bf16(false, a, false, b, (short)0, c, false, false);
}

// fp32x4 -> bf16x4 staging store (global_load_b128 -> 2x cvt_pk -> ds_store_b64)
__device__ __forceinline__ void stage16(bf16* __restrict__ dst, const float* __restrict__ src) {
#pragma unroll
  for (int j = 0; j < 4; ++j) {
    const float4 v = *(const float4*)(src + j * 4);
    v4bf b;
    b[0] = (bf16)v.x; b[1] = (bf16)v.y; b[2] = (bf16)v.z; b[3] = (bf16)v.w;
    *(v4bf*)(dst + j * 4) = b;
  }
}

// ---------------------------------------------------------------------------
// Kernel 1: router — logits, softmax, top-8, per-expert token lists
// ---------------------------------------------------------------------------
__global__ __launch_bounds__(256)
void moe_router(const float* __restrict__ x, const float* __restrict__ Wg,
                int* __restrict__ counts, int* __restrict__ tok_idx,
                float* __restrict__ tok_w, float* __restrict__ s_shared) {
  __shared__ float xrow[H_DIM];
  __shared__ float logits[E_EXP + 1];
  const int t   = blockIdx.x;
  const int tid = threadIdx.x;

  for (int i = tid; i < H_DIM; i += 256) xrow[i] = x[(size_t)t * H_DIM + i];
  __syncthreads();

  if (tid <= E_EXP) {
    float s = 0.f;
    for (int h = 0; h < H_DIM; ++h) s += xrow[h] * Wg[(size_t)h * (E_EXP + 1) + tid];
    logits[tid] = s;
  }
  __syncthreads();

  if (tid == 0) {
    float m = logits[0];
    for (int e = 1; e < E_EXP; ++e) m = fmaxf(m, logits[e]);
    float sum = 0.f;
    for (int e = 0; e < E_EXP; ++e) { float p = expf(logits[e] - m); xrow[e] = p; sum += p; }
    const float inv = 1.f / sum;
    int   sel[K_TOP];
    float w[K_TOP];
    float wsum = 0.f;
    for (int k = 0; k < K_TOP; ++k) {
      int best = 0; float bv = -1.f;
      for (int e = 0; e < E_EXP; ++e) {
        bool used = false;
        for (int j = 0; j < k; ++j) used |= (sel[j] == e);
        const float v = xrow[e];
        if (!used && v > bv) { bv = v; best = e; }
      }
      sel[k] = best; w[k] = bv * inv; wsum += bv * inv;
    }
    const float winv = 1.f / wsum;
    for (int k = 0; k < K_TOP; ++k) {
      const int e   = sel[k];
      const int pos = atomicAdd(&counts[e], 1);
      tok_idx[e * T_TOK + pos] = t;
      tok_w[e * T_TOK + pos]   = w[k] * winv;
    }
    s_shared[t] = 1.f / (1.f + expf(-logits[E_EXP]));
  }
}

// ---------------------------------------------------------------------------
// Kernel 2: fused expert FFN  out += rowscale * (silu(X Wg) * (X Wu)) Wd
// One workgroup (8 wave32) per 64-token tile of one expert (or shared expert).
// ---------------------------------------------------------------------------
#define TILE_M 64
#define XK 64      // K chunk staged per iteration (2 WMMA k-steps)
#define XS 72      // padded LDS stride for X chunk (bf16): 144B, conflict-free
#define XSZ (TILE_M * XS)
#define AS 520     // padded LDS stride for act tile (bf16)

__global__ __launch_bounds__(256)
void moe_ffn(const float* __restrict__ x,
             const float* __restrict__ Wgu,        // [H][2F] (per-expert when routed)
             const float* __restrict__ Wd,         // [F][H]
             const int*   __restrict__ tok_idx,    // routed: per-expert token lists
             const float* __restrict__ tok_w,      // routed: combine weights
             const float* __restrict__ s_shared,   // shared: sigmoid(gate)
             const int*   __restrict__ count_ptr,  // routed: tokens per expert
             float* __restrict__ out,
             int shared_mode) {
  __shared__ bf16  Xs[2 * XSZ];            // double-buffered X chunk (18 KB)
  __shared__ bf16  As_[TILE_M * AS];       // 65 KB act tile (bf16)
  __shared__ int   rowtok[TILE_M];
  __shared__ float rowscale[TILE_M];

  const int tid  = threadIdx.x;
  const int lane = tid & 31;
  const int wave = tid >> 5;               // 0..7
  const int e    = blockIdx.x;
  const int tile = blockIdx.y;

  int count;
  const float *wgu, *wd;
  if (shared_mode) {
    count = T_TOK;
    wgu = Wgu; wd = Wd;
  } else {
    count = count_ptr[e];
    wgu = Wgu + (size_t)e * H_DIM * (2 * F_DIM);
    wd  = Wd  + (size_t)e * F_DIM * H_DIM;
  }
  const int mbase = tile * TILE_M;
  if (mbase >= count) return;              // block-uniform early exit

  if (tid < TILE_M) {
    const int g = mbase + tid;
    if (shared_mode)      { rowtok[tid] = g;                       rowscale[tid] = s_shared[g]; }
    else if (g < count)   { rowtok[tid] = tok_idx[e * T_TOK + g];  rowscale[tid] = tok_w[e * T_TOK + g]; }
    else                  { rowtok[tid] = 0;                       rowscale[tid] = 0.f; }
  }
  __syncthreads();

  // Per-thread staging slice: 4 threads per row, 16 columns (4x float4) each.
  // Gather pointer resolved ONCE and kept in registers for the whole kernel.
  const int myrow = tid >> 2;                         // 0..63
  const int mycol = (tid & 3) * 16;                   // 0/16/32/48
  const float* __restrict__ px = x + (size_t)rowtok[myrow] * H_DIM + mycol;
  bf16* const xdst = Xs + myrow * XS + mycol;

  const int nlane = lane & 15;
  const int half  = lane >> 4;

  // ---------------- Phase 1: GU = X * Wgu ; act = silu(g)*u*scale -> LDS ----
  const int NIT = H_DIM / XK;                         // 32
  for (int pass = 0; pass < 4; ++pass) {
    const int cg = pass * 128 + wave * 16;            // gate column base (0..511)
    v8f accg[4] = {};
    v8f accu[4] = {};

    stage16(xdst, px);                                // prime buffer 0 (kc = 0)
    __syncthreads();

    for (int it = 0; it < NIT; ++it) {
      const bf16* xb = &Xs[(it & 1) * XSZ];
      if (it + 1 < NIT)                               // stage next chunk into other buffer
        stage16(xdst + ((it + 1) & 1) * XSZ, px + (it + 1) * XK);
      __builtin_prefetch(wgu + (size_t)(it * XK + XK) * (2 * F_DIM) + cg, 0, 0);
#pragma unroll
      for (int ks = 0; ks < XK; ks += 32) {
        const int kc = it * XK + ks;
        const v16bf bg = load_b_frag(wgu, 2 * F_DIM, kc, cg,         lane);
        const v16bf bu = load_b_frag(wgu, 2 * F_DIM, kc, cg + F_DIM, lane);
#pragma unroll
        for (int mt = 0; mt < 4; ++mt) {
          const v16bf a = load_a_frag(xb + (mt * 16) * XS + ks, XS, lane);
          accg[mt] = wmma_bf16(a, bg, accg[mt]);
          accu[mt] = wmma_bf16(a, bu, accu[mt]);
        }
      }
      __syncthreads();                                // staging done & reads done
    }

    // fused SiLU*up*combine epilogue -> bf16 act tile
#pragma unroll
    for (int mt = 0; mt < 4; ++mt) {
#pragma unroll
      for (int r = 0; r < 8; ++r) {
        const int row = mt * 16 + r + (half << 3);
        const float g = accg[mt][r];
        const float u = accu[mt][r];
        const float a = (g / (1.f + expf(-g))) * u * rowscale[row];
        As_[row * AS + (cg + nlane)] = (bf16)a;
      }
    }
  }
  __syncthreads();

  // ---------------- Phase 2: OUT += act * Wd (atomic fp32 scatter) ----------
  for (int nt = 0; nt < 16; ++nt) {
    const int nb = wave * 256 + nt * 16;
    v8f acc[4] = {};
    for (int kc = 0; kc < F_DIM; kc += 32) {
      if (kc + 32 < F_DIM)
        __builtin_prefetch(wd + (size_t)(kc + 32) * H_DIM + nb, 0, 0);
      const v16bf b = load_b_frag(wd, H_DIM, kc, nb, lane);
#pragma unroll
      for (int mt = 0; mt < 4; ++mt) {
        const v16bf a = load_a_frag(&As_[(mt * 16) * AS + kc], AS, lane);
        acc[mt] = wmma_bf16(a, b, acc[mt]);
      }
    }
#pragma unroll
    for (int mt = 0; mt < 4; ++mt) {
#pragma unroll
      for (int r = 0; r < 8; ++r) {
        const int row = mt * 16 + r + (half << 3);
        if (mbase + row < count) {
          atomicAdd(&out[(size_t)rowtok[row] * H_DIM + nb + nlane], acc[mt][r]);
        }
      }
    }
  }
}

// ---------------------------------------------------------------------------
// Launch
// ---------------------------------------------------------------------------
extern "C" void kernel_launch(void* const* d_in, const int* in_sizes, int n_in,
                              void* d_out, int out_size, void* d_ws, size_t ws_size,
                              hipStream_t stream) {
  const float* x    = (const float*)d_in[0];   // (T,H)
  const float* Wg   = (const float*)d_in[1];   // (H,E+1)
  const float* Wgu  = (const float*)d_in[2];   // (E,H,2F)
  const float* Wd   = (const float*)d_in[3];   // (E,F,H)
  const float* Wsgu = (const float*)d_in[4];   // (H,2FS)
  const float* Wsd  = (const float*)d_in[5];   // (FS,H)
  float* out = (float*)d_out;

  char* ws = (char*)d_ws;
  int*   counts  = (int*)ws;                                               // 64 ints
  int*   tok_idx = (int*)(ws + 1024);                                      // E*T ints
  float* tok_w   = (float*)(ws + 1024 + (size_t)E_EXP * T_TOK * 4);        // E*T floats
  float* s_sh    = (float*)(ws + 1024 + (size_t)2 * E_EXP * T_TOK * 4);    // T floats

  hipMemsetAsync(counts, 0, E_EXP * sizeof(int), stream);
  hipMemsetAsync(out, 0, (size_t)T_TOK * H_DIM * sizeof(float), stream);

  moe_router<<<T_TOK, 256, 0, stream>>>(x, Wg, counts, tok_idx, tok_w, s_sh);

  // Routed experts: grid.x = expert, grid.y = token tile (early-exit past count)
  moe_ffn<<<dim3(E_EXP, T_TOK / TILE_M), 256, 0, stream>>>(
      x, Wgu, Wd, tok_idx, tok_w, s_sh, counts, out, 0);

  // Shared expert: identical shapes (FS == F), sigmoid gate as row scale
  moe_ffn<<<dim3(1, T_TOK / TILE_M), 256, 0, stream>>>(
      x, Wsgu, Wsd, nullptr, nullptr, s_sh, nullptr, out, 1);
}
